// DeepEMD_CL_51084341019266
// MI455X (gfx1250) — matrix operand order repaired and checked by reading, not compiled, not run on hardware
//
#include <hip/hip_runtime.h>

typedef __attribute__((ext_vector_type(16))) _Float16 v16h;
typedef __attribute__((ext_vector_type(8)))  float    v8f;

#define WAY   16
#define NQ    1024
#define CCH   640
#define LL    49      // 7*7 nodes
#define KPITCH 51     // odd pitch -> conflict-free row & column LDS access

__device__ __forceinline__ int imin(int a, int b) { return a < b ? a : b; }

// ---------------------------------------------------------------------------
// Kernel 1: adaptive_avg_pool2d -> Qp[1024][640], Bp[16][640]
// ---------------------------------------------------------------------------
__global__ __launch_bounds__(256)
void pool_kernel(const float* __restrict__ query, const float* __restrict__ proto,
                 float* __restrict__ Qp, float* __restrict__ Bp) {
    int idx = blockIdx.x * 256 + threadIdx.x;
    const int nq = NQ * CCH;
    if (idx < nq) {
        const float* p = query + (size_t)idx * LL;
        float s = 0.f;
        #pragma unroll
        for (int k = 0; k < LL; ++k) s += p[k];
        Qp[idx] = s * (1.0f / 49.0f);
    } else if (idx < nq + WAY * CCH) {
        int j = idx - nq;
        const float* p = proto + (size_t)j * LL;
        float s = 0.f;
        #pragma unroll
        for (int k = 0; k < LL; ++k) s += p[k];
        Bp[j] = s * (1.0f / 49.0f);
    }
}

// ---------------------------------------------------------------------------
// Kernel 2: per-sample channel-center + L2-normalize per node; transpose to
// node-major f16: qn[s][l][c], pn[s][l][c]  (c contiguous -> WMMA fragments)
// blockIdx.x: 0..1023 = query samples, 1024..1039 = proto samples
// Uses Var trick: ||x-m||^2 = sum x^2 - (sum x)^2 / C  (single stats pass)
// ---------------------------------------------------------------------------
__global__ __launch_bounds__(256)
void norm_kernel(const float* __restrict__ query, const float* __restrict__ proto,
                 _Float16* __restrict__ qn, _Float16* __restrict__ pn) {
    int s = blockIdx.x;
    const float* src;
    _Float16* dst;
    if (s < NQ) { src = query + (size_t)s * CCH * LL; dst = qn + (size_t)s * LL * CCH; }
    else        { int p = s - NQ; src = proto + (size_t)p * CCH * LL; dst = pn + (size_t)p * LL * CCH; }

    __shared__ float tile[64][50];
    __shared__ float S1[LL], S2[LL], mean_[LL], inv_[LL];
    int tid = threadIdx.x;
    if (tid < LL) { S1[tid] = 0.f; S2[tid] = 0.f; }
    __syncthreads();

    for (int c0 = 0; c0 < CCH; c0 += 64) {
        for (int e = tid; e < 64 * LL; e += 256) {
            int cc = e / LL, l = e % LL;
            tile[cc][l] = src[(size_t)(c0 + cc) * LL + l];
        }
        __syncthreads();
        if (tid < LL) {
            float s1 = 0.f, s2 = 0.f;
            #pragma unroll 8
            for (int cc = 0; cc < 64; ++cc) { float x = tile[cc][tid]; s1 += x; s2 += x * x; }
            S1[tid] += s1; S2[tid] += s2;
        }
        __syncthreads();
    }
    if (tid < LL) {
        float m = S1[tid] * (1.0f / 640.0f);
        mean_[tid] = m;
        float nsq = S2[tid] - S1[tid] * m;           // centered sum of squares
        float nr = sqrtf(fmaxf(nsq, 0.f));
        inv_[tid] = 1.0f / fmaxf(nr, 1e-8f);         // COS_EPS
    }
    __syncthreads();
    // pass 2: write transposed, centered, normalized f16 (coalesced 128B rows)
    for (int c0 = 0; c0 < CCH; c0 += 64) {
        for (int e = tid; e < 64 * LL; e += 256) {
            int cc = e / LL, l = e % LL;
            tile[cc][l] = src[(size_t)(c0 + cc) * LL + l];
        }
        __syncthreads();
        for (int e = tid; e < LL * 64; e += 256) {
            int l = e >> 6, cc = e & 63;
            dst[(size_t)l * CCH + c0 + cc] = (_Float16)((tile[cc][l] - mean_[l]) * inv_[l]);
        }
        __syncthreads();
    }
}

// ---------------------------------------------------------------------------
// Kernel 3: a[q][n][l] = relu(sum_c query[q,c,l]*Bp[n,c]) + 1e-3, row-mass 49
// one block per q; query streamed through LDS once
// ---------------------------------------------------------------------------
__global__ __launch_bounds__(256)
void marg_a_kernel(const float* __restrict__ query, const float* __restrict__ Bp,
                   float* __restrict__ a_arr) {
    int q = blockIdx.x;
    __shared__ float qc[64][50];
    __shared__ float BpL[WAY][64];
    __shared__ float w[WAY * LL];
    __shared__ float rs[WAY];
    int tid = threadIdx.x;
    float acc[4] = {0.f, 0.f, 0.f, 0.f};

    for (int c0 = 0; c0 < CCH; c0 += 64) {
        for (int e = tid; e < 64 * LL; e += 256) {
            int cc = e / LL, l = e % LL;
            qc[cc][l] = query[(size_t)q * CCH * LL + (size_t)(c0 + cc) * LL + l];
        }
        for (int e = tid; e < WAY * 64; e += 256) {
            int nn = e >> 6, cc = e & 63;
            BpL[nn][cc] = Bp[nn * CCH + c0 + cc];
        }
        __syncthreads();
        #pragma unroll
        for (int k = 0; k < 4; ++k) {
            int idx = tid + k * 256;
            if (idx < WAY * LL) {
                int nn = idx / LL, l = idx % LL;
                float s = 0.f;
                #pragma unroll 8
                for (int cc = 0; cc < 64; ++cc) s += qc[cc][l] * BpL[nn][cc];
                acc[k] += s;
            }
        }
        __syncthreads();
    }
    for (int k = 0; k < 4; ++k) {
        int idx = tid + k * 256;
        if (idx < WAY * LL) w[idx] = fmaxf(acc[k], 0.f) + 1e-3f;
    }
    __syncthreads();
    if (tid < WAY) {
        float s = 0.f;
        for (int l = 0; l < LL; ++l) s += w[tid * LL + l];
        rs[tid] = 49.0f / s;
    }
    __syncthreads();
    for (int k = 0; k < 4; ++k) {
        int idx = tid + k * 256;
        if (idx < WAY * LL) a_arr[(size_t)q * (WAY * LL) + idx] = w[idx] * rs[idx / LL];
    }
}

// ---------------------------------------------------------------------------
// Kernel 4: b[q][n][l] = relu(sum_c proto[n,c,l]*Qp[q,c]) + 1e-3, mass 49
// block handles 4 q's (proto stays L2-resident; 512MB L2 traffic total)
// ---------------------------------------------------------------------------
__global__ __launch_bounds__(256)
void marg_b_kernel(const float* __restrict__ proto, const float* __restrict__ Qp,
                   float* __restrict__ b_arr) {
    int qg = blockIdx.x;            // 256 groups of 4 queries
    int tid = threadIdx.x;
    __shared__ float Qs[4][CCH];
    __shared__ float w[4][WAY * LL];
    __shared__ float rs[64];

    for (int e = tid; e < 4 * CCH; e += 256) Qs[e / CCH][e % CCH] = Qp[(size_t)qg * 4 * CCH + e];
    __syncthreads();

    float acc[4][4] = {};
    int nn_[4], l_[4];
    bool valid[4];
    #pragma unroll
    for (int k = 0; k < 4; ++k) {
        int idx = tid + k * 256;
        valid[k] = idx < WAY * LL;
        nn_[k] = valid[k] ? idx / LL : 0;
        l_[k]  = valid[k] ? idx % LL : 0;
    }
    for (int c = 0; c < CCH; ++c) {
        float pv[4];
        #pragma unroll
        for (int k = 0; k < 4; ++k) pv[k] = proto[((size_t)nn_[k] * CCH + c) * LL + l_[k]];
        #pragma unroll
        for (int k = 0; k < 4; ++k)
            #pragma unroll
            for (int qq = 0; qq < 4; ++qq) acc[k][qq] += pv[k] * Qs[qq][c];
    }
    for (int qq = 0; qq < 4; ++qq)
        for (int k = 0; k < 4; ++k) {
            int idx = tid + k * 256;
            if (idx < WAY * LL) w[qq][idx] = fmaxf(acc[k][qq], 0.f) + 1e-3f;
        }
    __syncthreads();
    if (tid < 64) {
        int qq = tid >> 4, nn = tid & 15;
        float s = 0.f;
        for (int l = 0; l < LL; ++l) s += w[qq][nn * LL + l];
        rs[tid] = 49.0f / s;
    }
    __syncthreads();
    for (int qq = 0; qq < 4; ++qq)
        for (int k = 0; k < 4; ++k) {
            int idx = tid + k * 256;
            if (idx < WAY * LL)
                b_arr[(size_t)(qg * 4 + qq) * (WAY * LL) + idx] = w[qq][idx] * rs[qq * 16 + idx / LL];
        }
}

// ---------------------------------------------------------------------------
// Kernel 5 (main): per (q,n) pair: WMMA cosine GEMM -> 49x49 sim in LDS,
// K = exp((sim-1)/eps) once, 50 regular-domain Sinkhorn iterations (pure FMA
// matvecs in LDS), logits = (T/L) * u^T (K .* sim) v.
// grid = (16, 1024), block = 256 (8 waves, 2 WMMA tiles each)
// ---------------------------------------------------------------------------
__global__ __launch_bounds__(256)
void emd_main_kernel(const _Float16* __restrict__ qn, const _Float16* __restrict__ pn,
                     const float* __restrict__ a_arr, const float* __restrict__ b_arr,
                     float* __restrict__ out) {
    int n = blockIdx.x;
    int q = blockIdx.y;
    int tid = threadIdx.x;

    __shared__ float S[64][66];                 // sim tile (padded 49->64)
    __shared__ float Kl[LL * KPITCH];           // Gibbs kernel
    __shared__ float av[LL], bv[LL];
    __shared__ float uu[64], vv[64], red[64];

    // ---- WMMA GEMM: sim[i][j] = sum_c qn[q,i,c] * pn[n,j,c], K = 640 ----
    int wave = tid >> 5, lane = tid & 31;
    int half = lane >> 4, l16 = lane & 15;
    int ti  = wave >> 1;                        // 0..3
    int i0  = ti * 16;
    int tj0 = (wave & 1) * 32;                  // this wave owns cols tj0..tj0+31

    int arow  = imin(i0 + l16, 48);             // pad rows clamp to last node
    int bcol0 = imin(tj0 + l16, 48);
    int bcol1 = imin(tj0 + 16 + l16, 48);
    const _Float16* ap  = qn + ((size_t)q * LL + arow)  * CCH;
    const _Float16* bp0 = pn + ((size_t)n * LL + bcol0) * CCH;
    const _Float16* bp1 = pn + ((size_t)n * LL + bcol1) * CCH;

    v8f acc0 = {0.f, 0.f, 0.f, 0.f, 0.f, 0.f, 0.f, 0.f};
    v8f acc1 = {0.f, 0.f, 0.f, 0.f, 0.f, 0.f, 0.f, 0.f};

    for (int k0 = 0; k0 < CCH; k0 += 32) {
        union { v16h v; float4 f[2]; } A, B0, B1;
        // A fragment (16x32 f16): lane holds row lane%16, K = half*16 .. +15
        A.f[0]  = *reinterpret_cast<const float4*>(ap  + k0 + half * 16);
        A.f[1]  = *reinterpret_cast<const float4*>(ap  + k0 + half * 16 + 8);
        // B fragment (32x16 f16): lane holds column lane%16, K = half*16 .. +15
        B0.f[0] = *reinterpret_cast<const float4*>(bp0 + k0 + half * 16);
        B0.f[1] = *reinterpret_cast<const float4*>(bp0 + k0 + half * 16 + 8);
        B1.f[0] = *reinterpret_cast<const float4*>(bp1 + k0 + half * 16);
        B1.f[1] = *reinterpret_cast<const float4*>(bp1 + k0 + half * 16 + 8);

        acc0 = __builtin_amdgcn_wmma_f32_16x16x32_f16(false, A.v, false, B0.v,
                                                      (short)0, acc0, false, false);
        acc1 = __builtin_amdgcn_wmma_f32_16x16x32_f16(false, A.v, false, B1.v,
                                                      (short)0, acc1, false, false);
    }
    // D layout: VGPR r -> M = r + 8*half, N = lane%16
    #pragma unroll
    for (int r = 0; r < 8; ++r) {
        int i = i0 + r + half * 8;
        S[i][tj0 + l16]      = acc0[r];
        S[i][tj0 + 16 + l16] = acc1[r];
    }
    __syncthreads();

    // ---- Gibbs kernel: K = exp(-cost/eps) = exp(20*(sim-1)) ----
    for (int e = tid; e < LL * LL; e += 256) {
        int i = e / LL, j = e % LL;
        Kl[i * KPITCH + j] = __expf((S[i][j] - 1.0f) * 20.0f);
    }
    if (tid < LL) {
        int base = (q * WAY + n) * LL;
        av[tid] = a_arr[base + tid];
        bv[tid] = b_arr[base + tid];
        vv[tid] = 1.0f;                         // g = 0  -> v = 1
    }
    __syncthreads();

    // ---- 50 Sinkhorn iterations, regular domain (u=e^f, v=e^g) ----
    for (int it = 0; it < 50; ++it) {
        if (tid < LL) {
            const float* kr = &Kl[tid * KPITCH];
            float s = 0.f;
            #pragma unroll 7
            for (int j = 0; j < LL; ++j) s += kr[j] * vv[j];
            uu[tid] = av[tid] / s;
        }
        __syncthreads();
        if (tid < LL) {
            float s = 0.f;
            #pragma unroll 7
            for (int i = 0; i < LL; ++i) s += Kl[i * KPITCH + tid] * uu[i];
            vv[tid] = bv[tid] / s;
        }
        __syncthreads();
    }

    // ---- logits = (T/L) * sum_ij sim * u_i K_ij v_j ----
    if (tid < 64) red[tid] = 0.f;
    __syncthreads();
    if (tid < LL) {
        const float* kr = &Kl[tid * KPITCH];
        float s = 0.f;
        #pragma unroll 7
        for (int j = 0; j < LL; ++j) s += kr[j] * S[tid][j] * vv[j];
        red[tid] = uu[tid] * s;
    }
    __syncthreads();
    if (tid == 0) {
        float t = 0.f;
        for (int i = 0; i < LL; ++i) t += red[i];
        out[q * WAY + n] = t * (12.5f / 49.0f);
    }
}

// ---------------------------------------------------------------------------
// launch: proto = d_in[0] (16*640*49 f32), query = d_in[1] (1024*640*49 f32)
// d_out = logits [1024,16] f32. Workspace ~74.3 MB.
// ---------------------------------------------------------------------------
extern "C" void kernel_launch(void* const* d_in, const int* in_sizes, int n_in,
                              void* d_out, int out_size, void* d_ws, size_t ws_size,
                              hipStream_t stream) {
    const float* proto = (const float*)d_in[0];
    const float* query = (const float*)d_in[1];
    char* ws = (char*)d_ws;

    float*    Qp    = (float*)(ws + 0);               //  2,621,440 B
    float*    Bp    = (float*)(ws + 2621440);         //     40,960 B
    _Float16* qn    = (_Float16*)(ws + 2662400);      // 64,225,280 B
    _Float16* pn    = (_Float16*)(ws + 66887680);     //  1,003,520 B
    float*    a_arr = (float*)(ws + 67891200);        //  3,211,264 B
    float*    b_arr = (float*)(ws + 71102464);        //  3,211,264 B -> 74,313,728 total

    pool_kernel  <<<2600, 256, 0, stream>>>(query, proto, Qp, Bp);
    norm_kernel  <<<1040, 256, 0, stream>>>(query, proto, qn, pn);
    marg_a_kernel<<<1024, 256, 0, stream>>>(query, Bp, a_arr);
    marg_b_kernel<<<256,  256, 0, stream>>>(proto, Qp, b_arr);
    emd_main_kernel<<<dim3(WAY, NQ), 256, 0, stream>>>(qn, pn, a_arr, b_arr, (float*)d_out);
}
